// MultiHeadAttention_11038065951497
// MI455X (gfx1250) — matrix operand order repaired and checked by reading, not compiled, not run on hardware
//
#include <hip/hip_runtime.h>
#include <hip/hip_bf16.h>

// ---------------------------------------------------------------------------
// MI455X (gfx1250) multi-head attention forward, wave32 + WMMA f16 (f32 acc).
// Single-pass attention: per-block 32x2048 score strip lives in LDS (257KB of
// the 320KB/WGP), softmax in LDS, attention written to HBM exactly once,
// context GEMM fed from f16-packed probabilities in LDS.
// GEMMs use kk-outer/j-inner register blocking with double-buffered B
// fragments so WMMAs overlap the next tile's global loads.
// ---------------------------------------------------------------------------

typedef __attribute__((ext_vector_type(16))) _Float16 v16h;
typedef __attribute__((ext_vector_type(4)))  _Float16 v4h;
typedef __attribute__((ext_vector_type(8)))  float    v8f;

#define D_   512
#define H_   8
#define DH_  64
#define S_   2048
#define B_   2
#define ROWS 4096              // B*S
#define LSTR 2052              // LDS stride (floats) for 2048-wide score rows
#define YSTR 516               // LDS stride (floats) for 512-wide proj rows

union HF16 { v16h v; float4 f4[2]; };

__device__ __forceinline__ v8f wmma32(v16h a, v16h b, v8f c) {
  // D = A(16x32 f16) * B(32x16 f16) + C(16x16 f32)
  return __builtin_amdgcn_wmma_f32_16x16x32_f16(false, a, false, b,
                                                (short)0, c, false, false);
}

// A fragment (16x32 f16). ISA layout: lane L<16 -> row L, K{0..7,16..23};
// lane L>=16 -> row L-16, K{8..15,24..31}. base = tile origin (row0, k0).
// Works for global f16 and for LDS-resident packed f16 (generic pointer).
__device__ __forceinline__ v16h load_a_f16(const _Float16* base, int ld) {
  const int lane = threadIdx.x & 31;
  const int row  = lane & 15;
  const int g8   = (lane >> 4) << 3;      // 0 or 8
  const _Float16* p = base + row * ld + g8;
  HF16 u;
  u.f4[0] = *(const float4*)(p);          // K = g8 .. g8+7
  u.f4[1] = *(const float4*)(p + 16);     // K = 16+g8 .. 16+g8+7
  return u.v;
}

// B fragment (32x16 f16). Lane n<16 holds column n, K{0..7,16..23}; lane n+16
// holds column n, K{8..15,24..31}. Column n of B is a contiguous memory row here.
__device__ __forceinline__ v16h load_b_f16(const _Float16* base, int ld) {
  const int lane = threadIdx.x & 31;
  const int n    = lane & 15;
  const int g16  = (lane >> 4) << 3;      // 0 or 8
  const _Float16* p = base + n * ld + g16;
  HF16 u;
  u.f4[0] = *(const float4*)(p);
  u.f4[1] = *(const float4*)(p + 16);
  return u.v;
}

// ---------------------------------------------------------------------------
// K0: f32 -> f16 conversion (inputs + weights), 4-wide.
// ---------------------------------------------------------------------------
__global__ void cvt4_kernel(const float* __restrict__ s,
                            _Float16* __restrict__ d, int n4) {
  int i = blockIdx.x * blockDim.x + threadIdx.x;
  if (i < n4) {
    float4 v = ((const float4*)s)[i];
    v4h o = { (_Float16)v.x, (_Float16)v.y, (_Float16)v.z, (_Float16)v.w };
    ((v4h*)d)[i] = o;
  }
}

// ---------------------------------------------------------------------------
// K1: Q/K/V projections. y = x @ W^T. 256 thr = 8 waves.
// kk-outer / j-inner register blocking, acc[8] block, double-buffered B:
// the load for tile j+1 issues before the WMMA for tile j.
// Q,K stored [bh][s][dh] f16; V stored transposed [bh][dh][s] f16.
// ---------------------------------------------------------------------------
__global__ void qkv_proj_kernel(const _Float16* __restrict__ xq,
                                const _Float16* __restrict__ xk,
                                const _Float16* __restrict__ xv,
                                const _Float16* __restrict__ wq,
                                const _Float16* __restrict__ wk,
                                const _Float16* __restrict__ wv,
                                _Float16* __restrict__ Qo,
                                _Float16* __restrict__ Ko,
                                _Float16* __restrict__ Vt) {
  const int which = blockIdx.x >> 7;        // 0=q 1=k 2=v (128 row-blocks each)
  const int rb    = blockIdx.x & 127;
  const _Float16* x = (which == 0) ? xq : (which == 1) ? xk : xv;
  const _Float16* w = (which == 0) ? wq : (which == 1) ? wk : wv;
  const int wave = threadIdx.x >> 5;
  const int lane = threadIdx.x & 31;
  const int mt   = wave & 1;
  const int m0   = rb * 32 + mt * 16;
  const int ntb  = (wave >> 1);             // n-tile base, stride 4

  v8f acc[8];
#pragma unroll
  for (int j = 0; j < 8; ++j) acc[j] = (v8f){};

  for (int kk = 0; kk < 16; ++kk) {         // K = 512 = 16 x 32
    v16h a = load_a_f16(x + m0 * D_ + kk * 32, D_);
    v16h bcur = load_b_f16(w + (size_t)(ntb * 16) * D_ + kk * 32, D_);
#pragma unroll
    for (int j = 0; j < 8; ++j) {
      v16h bnext;
      if (j < 7)
        bnext = load_b_f16(w + (size_t)((ntb + (j + 1) * 4) * 16) * D_ + kk * 32,
                           D_);
      acc[j] = wmma32(a, bcur, acc[j]);     // overlaps bnext's loads
      bcur = bnext;
    }
  }

#pragma unroll
  for (int j = 0; j < 8; ++j) {
    const int col = (ntb + j * 4) * 16 + (lane & 15);
    const int h = col >> 6, dh = col & 63;
#pragma unroll
    for (int i = 0; i < 8; ++i) {
      const int m = m0 + i + ((lane >> 4) << 3);
      const int bb = m >> 11, s = m & (S_ - 1);
      const int bh = bb * H_ + h;
      _Float16 val = (_Float16)acc[j][i];
      if (which == 0)      Qo[(bh * S_ + s) * DH_ + dh] = val;
      else if (which == 1) Ko[(bh * S_ + s) * DH_ + dh] = val;
      else                 Vt[(bh * DH_ + dh) * S_ + s] = val;
    }
  }
}

// ---------------------------------------------------------------------------
// K2: fused scores -> softmax -> attention write -> context.
// grid = 16 heads * 64 row-blocks. Dynamic LDS: 32 x LSTR f32 (~257 KB).
// ---------------------------------------------------------------------------
__global__ void attn_kernel(const _Float16* __restrict__ Q,
                            const _Float16* __restrict__ K,
                            const _Float16* __restrict__ Vt,
                            _Float16* __restrict__ ctx,
                            float* __restrict__ attn_out) {
  extern __shared__ float Sld[];              // [32][LSTR]
  const int bh = blockIdx.x >> 6;
  const int rb = blockIdx.x & 63;
  const int m0 = rb * 32;
  const int wave = threadIdx.x >> 5;
  const int lane = threadIdx.x & 31;
  const _Float16* Qh = Q  + (size_t)bh * S_ * DH_;
  const _Float16* Kh = K  + (size_t)bh * S_ * DH_;
  const _Float16* Vh = Vt + (size_t)bh * DH_ * S_;
  const float scale = 0.125f;                 // DH^-0.5

  // ---- phase 1: scores S[32][2048] = Q Kt * scale into LDS ----
  {
    const int mt = wave & 1;
    const v16h aq0 = load_a_f16(Qh + (m0 + mt * 16) * DH_,      DH_);
    const v16h aq1 = load_a_f16(Qh + (m0 + mt * 16) * DH_ + 32, DH_);
    for (int jj = 0; jj < 8; ++jj) {
      const int ntb = (wave >> 1) + jj * 16;  // tiles ntb + {0,4,8,12}
      // prefetch next group's K rows (global_prefetch_b8)
      if (jj < 7) {
        const _Float16* pf = Kh + (size_t)(ntb + 16 + (lane >> 3) * 4) * 16 * DH_
                                + (lane & 7) * 128;
        __builtin_prefetch((const void*)pf, 0, 1);
      }
      v16h b[4][2];
#pragma unroll
      for (int t = 0; t < 4; ++t) {
        const _Float16* kb = Kh + (size_t)(ntb + t * 4) * 16 * DH_;
        b[t][0] = load_b_f16(kb,      DH_);
        b[t][1] = load_b_f16(kb + 32, DH_);
      }
      v8f acc[4];
#pragma unroll
      for (int t = 0; t < 4; ++t) {
        acc[t] = wmma32(aq0, b[t][0], (v8f){});
        acc[t] = wmma32(aq1, b[t][1], acc[t]);
      }
#pragma unroll
      for (int t = 0; t < 4; ++t) {
        const int col = (ntb + t * 4) * 16 + (lane & 15);
#pragma unroll
        for (int i = 0; i < 8; ++i) {
          const int r = mt * 16 + i + ((lane >> 4) << 3);
          Sld[r * LSTR + col] = acc[t][i] * scale;
        }
      }
    }
  }
  __syncthreads();

  // ---- phase 2: exact softmax per row; write f32 attention to HBM once,
  //      and pack normalized probs as f16 in place (front half of each row)
  //      for the context GEMM. Each row is owned by exactly one wave; the
  //      f32 reads precede the packed f16 writes in program order. ----
  for (int rr = 0; rr < 4; ++rr) {
    const int r = wave * 4 + rr;
    float* row = Sld + r * LSTR;
    float mx = -3.4e38f;
    for (int c = lane; c < S_; c += 32) mx = fmaxf(mx, row[c]);
#pragma unroll
    for (int off = 16; off; off >>= 1) mx = fmaxf(mx, __shfl_xor(mx, off));
    float sum = 0.f;
    for (int c = lane; c < S_; c += 32) {
      float e = __expf(row[c] - mx);
      row[c] = e;
      sum += e;
    }
#pragma unroll
    for (int off = 16; off; off >>= 1) sum += __shfl_xor(sum, off);
    const float inv = 1.0f / sum;
    float* gout = attn_out + ((long)(bh * S_ + m0 + r)) * S_;
    _Float16* rowh = (_Float16*)row;          // packed f16 view of same row
    for (int c4 = lane * 4; c4 < S_; c4 += 128) {
      float4 v = *(float4*)(row + c4);        // ds_load_b128 (before store)
      v.x *= inv; v.y *= inv; v.z *= inv; v.w *= inv;
      *(float4*)(gout + c4) = v;              // one-time HBM write
      v4h pk = { (_Float16)v.x, (_Float16)v.y, (_Float16)v.z, (_Float16)v.w };
      *(v4h*)(rowh + c4) = pk;                // ds_store_b64, packed probs
    }
  }
  __syncthreads();

  // ---- phase 3: context C[32][64] = A[32][2048] @ V[2048][64].
  //      A fragments read as packed f16 straight from LDS (ds_load_b128),
  //      two interleaved accumulator chains to break WMMA latency. ----
  {
    const int mt = wave & 1;
    const int dt = wave >> 1;                 // 0..3 -> 16 dh cols each
    const _Float16* Ald = (const _Float16*)(Sld + (size_t)(mt * 16) * LSTR);
    const _Float16* Vb  = Vh + (size_t)(dt * 16) * S_;
    v8f acc0 = {}, acc1 = {};
    for (int kk = 0; kk < 64; kk += 2) {      // K = 2048 = 64 x 32
      v16h a0 = load_a_f16(Ald + kk * 32,      LSTR * 2);
      v16h b0 = load_b_f16(Vb  + kk * 32,      S_);
      acc0 = wmma32(a0, b0, acc0);
      v16h a1 = load_a_f16(Ald + kk * 32 + 32, LSTR * 2);
      v16h b1 = load_b_f16(Vb  + kk * 32 + 32, S_);
      acc1 = wmma32(a1, b1, acc1);
    }
    const int col = dt * 16 + (lane & 15);
    const int hh = bh & (H_ - 1), bb = bh >> 3;
#pragma unroll
    for (int i = 0; i < 8; ++i) {
      const int s = m0 + mt * 16 + i + ((lane >> 4) << 3);
      ctx[((long)(bb * S_ + s)) * D_ + hh * DH_ + col] =
          (_Float16)(acc0[i] + acc1[i]);
    }
  }
}

// ---------------------------------------------------------------------------
// K3: output projection + bias + residual + LayerNorm. grid = 128 row-blocks.
// Same register blocking + double-buffered B as K1.
// ---------------------------------------------------------------------------
__global__ void out_ln_kernel(const _Float16* __restrict__ ctx,
                              const _Float16* __restrict__ wo,
                              const float* __restrict__ bo,
                              const float* __restrict__ resid,
                              const float* __restrict__ gamma,
                              const float* __restrict__ beta,
                              float* __restrict__ out) {
  extern __shared__ float Yld[];              // [32][YSTR]
  const int rb = blockIdx.x;
  const int wave = threadIdx.x >> 5;
  const int lane = threadIdx.x & 31;
  const int mt = wave & 1;
  const int m0 = rb * 32 + mt * 16;
  const int ntb = (wave >> 1);

  v8f acc[8];
#pragma unroll
  for (int j = 0; j < 8; ++j) acc[j] = (v8f){};

  for (int kk = 0; kk < 16; ++kk) {
    v16h a = load_a_f16(ctx + m0 * D_ + kk * 32, D_);
    v16h bcur = load_b_f16(wo + (size_t)(ntb * 16) * D_ + kk * 32, D_);
#pragma unroll
    for (int j = 0; j < 8; ++j) {
      v16h bnext;
      if (j < 7)
        bnext = load_b_f16(wo + (size_t)((ntb + (j + 1) * 4) * 16) * D_ + kk * 32,
                           D_);
      acc[j] = wmma32(a, bcur, acc[j]);
      bcur = bnext;
    }
  }

#pragma unroll
  for (int j = 0; j < 8; ++j) {
    const int col = (ntb + j * 4) * 16 + (lane & 15);
#pragma unroll
    for (int i = 0; i < 8; ++i) {
      const int r = mt * 16 + i + ((lane >> 4) << 3);
      Yld[r * YSTR + col] = acc[j][i];
    }
  }
  __syncthreads();

  for (int rr = 0; rr < 4; ++rr) {
    const int r = wave * 4 + rr;
    const int gm = rb * 32 + r;
    float* row = Yld + r * YSTR;
    float sum = 0.f, sq = 0.f;
    for (int c = lane; c < D_; c += 32) {
      float v = row[c] + bo[c] + resid[gm * D_ + c];
      row[c] = v;
      sum += v; sq += v * v;
    }
#pragma unroll
    for (int off = 16; off; off >>= 1) {
      sum += __shfl_xor(sum, off);
      sq  += __shfl_xor(sq,  off);
    }
    const float mean = sum * (1.0f / D_);
    const float var  = sq * (1.0f / D_) - mean * mean;
    const float rstd = rsqrtf(var + 1e-5f);
    for (int c = lane; c < D_; c += 32)
      out[gm * D_ + c] = (row[c] - mean) * rstd * gamma[c] + beta[c];
  }
}

// ---------------------------------------------------------------------------
extern "C" void kernel_launch(void* const* d_in, const int* in_sizes, int n_in,
                              void* d_out, int out_size, void* d_ws,
                              size_t ws_size, hipStream_t stream) {
  (void)in_sizes; (void)n_in; (void)out_size; (void)ws_size;
  const float* query = (const float*)d_in[0];
  const float* key_  = (const float*)d_in[1];
  const float* value = (const float*)d_in[2];
  const float* Wq    = (const float*)d_in[3];
  const float* Wk    = (const float*)d_in[4];
  const float* Wv    = (const float*)d_in[5];
  const float* Wo    = (const float*)d_in[6];
  const float* bo    = (const float*)d_in[7];
  const float* gamma = (const float*)d_in[8];
  const float* beta  = (const float*)d_in[9];

  float* out  = (float*)d_out;                 // [B,S,D] = 2,097,152 f32
  float* attn = out + (size_t)ROWS * D_;       // [B*H,S,S] f32

  // Workspace layout (f16), ~31.5 MB total.
  _Float16* w = (_Float16*)d_ws;
  const size_t XSZ = (size_t)ROWS * D_;        // 2,097,152
  const size_t WSZ = (size_t)D_ * D_;          // 262,144
  _Float16 *xq16 = w,            *xk16 = w + XSZ,      *xv16 = w + 2 * XSZ;
  _Float16 *wq16 = w + 3 * XSZ,  *wk16 = wq16 + WSZ,   *wv16 = wk16 + WSZ,
           *wo16 = wv16 + WSZ;
  _Float16 *Q16  = wo16 + WSZ,   *K16 = Q16 + XSZ,     *Vt16 = K16 + XSZ,
           *ctx16 = Vt16 + XSZ;

  // K0: convert activations + weights to f16
  auto cvt = [&](const float* s, _Float16* d, size_t n) {
    int n4 = (int)(n / 4);
    cvt4_kernel<<<(n4 + 255) / 256, 256, 0, stream>>>(s, d, n4);
  };
  cvt(query, xq16, XSZ);
  cvt(key_,  xk16, XSZ);
  cvt(value, xv16, XSZ);
  cvt(Wq, wq16, WSZ);
  cvt(Wk, wk16, WSZ);
  cvt(Wv, wv16, WSZ);
  cvt(Wo, wo16, WSZ);

  // K1: Q/K/V projections (V transposed per head)
  qkv_proj_kernel<<<3 * 128, 256, 0, stream>>>(xq16, xk16, xv16,
                                               wq16, wk16, wv16,
                                               Q16, K16, Vt16);

  // K2: fused scores/softmax/attention-write/context; 257 KB LDS per WG
  attn_kernel<<<16 * 64, 256, 32 * LSTR * sizeof(float), stream>>>(
      Q16, K16, Vt16, ctx16, attn);

  // K3: output projection + residual + LayerNorm
  out_ln_kernel<<<128, 256, 32 * YSTR * sizeof(float), stream>>>(
      ctx16, wo16, bo, query, gamma, beta, out);
}
